// HashGenerator_52553219834302
// MI455X (gfx1250) — matrix operand order repaired and checked by reading, not compiled, not run on hardware
//
#include <hip/hip_runtime.h>
#include <hip/hip_bf16.h>

typedef __attribute__((ext_vector_type(16))) _Float16 v16h;
typedef __attribute__((ext_vector_type(8)))  _Float16 v8h;
typedef __attribute__((ext_vector_type(2)))  _Float16 v2h;
typedef __attribute__((ext_vector_type(8)))  float    v8f;

#define BATCH 8
#define ZDIM 512
#define SDIM 256
#define LVLS 16
#define TSIZE 16384
#define FEAT 32       // LVLS * 2
#define HID 64
#define NPIX 65536
#define SQRT2F 1.41421356237309515f

// per-batch packed weight blob (halves): [A0: 2048][A1: 4096][A2: 1024]
#define W_A0_OFF 0
#define W_A1_OFF 2048
#define W_A2_OFF 6144
#define W_TOTAL  7168

__device__ __forceinline__ float lrelu(float x) {
    return (x < 0.f ? 0.2f * x : x) * SQRT2F;
}

// ---------------------------------------------------------------------------
// Kernel 1: style mapping  s = lrelu(lrelu(lrelu(z@w1+b1)@w2+b2)@w3+b3)
// ---------------------------------------------------------------------------
__global__ __launch_bounds__(256) void style_kernel(
    const float* __restrict__ z,
    const float* __restrict__ w1, const float* __restrict__ b1,
    const float* __restrict__ w2, const float* __restrict__ b2,
    const float* __restrict__ w3, const float* __restrict__ b3,
    float* __restrict__ s_out)  // [BATCH][SDIM]
{
    __shared__ float smA[ZDIM];
    __shared__ float smB[SDIM];
    const int b = blockIdx.x;
    const int t = threadIdx.x;

    for (int i = t; i < ZDIM; i += 256) smA[i] = z[b * ZDIM + i];
    __syncthreads();

    float acc = b1[t];
    for (int i = 0; i < ZDIM; ++i) acc += smA[i] * w1[i * SDIM + t];
    smB[t] = lrelu(acc);
    __syncthreads();

    acc = b2[t];
    for (int i = 0; i < SDIM; ++i) acc += smB[i] * w2[i * SDIM + t];
    smA[t] = lrelu(acc);
    __syncthreads();

    acc = b3[t];
    for (int i = 0; i < SDIM; ++i) acc += smA[i] * w3[i * SDIM + t];
    s_out[b * SDIM + t] = lrelu(acc);
}

// ---------------------------------------------------------------------------
// Kernel 2: per-sample prep.  Produces one packed f16 weight blob per sample,
// pre-swizzled into the WMMA A-operand lane layout (transposed orientation:
// A rows = output channel o, A cols = input k):
//    element (lane,e):  o = tile*16 + (lane&15)
//                       k = kchunk*32 + (lane>>4)*16 + e
// Demodulation d[o] folded into weights; hash-table modulation mod[b][k]
// folded into the layer-0 weights ((x*mod)@W == x@(mod*W), mod is per-k).
// ---------------------------------------------------------------------------
__global__ __launch_bounds__(256) void prep_kernel(
    const float* __restrict__ s_in,      // [BATCH][SDIM]
    const float* __restrict__ gen_w,     // [LVLS][SDIM][2]
    const float* __restrict__ a0_w, const float* __restrict__ a0_b,
    const float* __restrict__ m0_w,      // [32][64]
    const float* __restrict__ a1_w, const float* __restrict__ a1_b,
    const float* __restrict__ m1_w,      // [64][64]
    const float* __restrict__ a2_w, const float* __restrict__ a2_b,
    const float* __restrict__ m2_w,      // [64][3]
    _Float16* __restrict__ Wout)         // [BATCH][W_TOTAL]
{
    __shared__ float ssm[SDIM];
    __shared__ float smod[FEAT];
    __shared__ float sc[HID];
    __shared__ float dd[HID];
    const int b = blockIdx.x;
    const int t = threadIdx.x;
    _Float16* __restrict__ Wb = Wout + b * W_TOTAL;

    ssm[t] = s_in[b * SDIM + t];
    __syncthreads();

    // --- style->table modulation scalars, kept in LDS (index k = 2*lvl+f) ---
    if (t < 32) {
        int l = t >> 1, f = t & 1;
        float a = 0.f;
        for (int j = 0; j < SDIM; ++j) a += ssm[j] * gen_w[(l * SDIM + j) * 2 + f];
        smod[t] = 1.f + a;
    }

    // --- layer 0 : in=32, out=64 (mod folded in) ---
    if (t < 32) {
        float a = a0_b[t];
        for (int j = 0; j < SDIM; ++j) a += ssm[j] * a0_w[j * 32 + t];
        sc[t] = a;
    }
    __syncthreads();
    if (t < 64) {
        float q = 0.f;
        for (int k = 0; k < 32; ++k) { float w = m0_w[k * 64 + t] * sc[k]; q += w * w; }
        dd[t] = rsqrtf(q + 1e-8f);
    }
    __syncthreads();
    for (int idx = t; idx < 4 * 32 * 16; idx += 256) {
        int n = idx >> 9, rem = idx & 511, lane = rem >> 4, e = rem & 15;
        int o = n * 16 + (lane & 15);
        int k = ((lane >> 4) << 4) + e;
        Wb[W_A0_OFF + idx] = (_Float16)(m0_w[k * 64 + o] * sc[k] * dd[o] * smod[k]);
    }
    __syncthreads();

    // --- layer 1 : in=64, out=64 ---
    if (t < 64) {
        float a = a1_b[t];
        for (int j = 0; j < SDIM; ++j) a += ssm[j] * a1_w[j * 64 + t];
        sc[t] = a;
    }
    __syncthreads();
    if (t < 64) {
        float q = 0.f;
        for (int k = 0; k < 64; ++k) { float w = m1_w[k * 64 + t] * sc[k]; q += w * w; }
        dd[t] = rsqrtf(q + 1e-8f);
    }
    __syncthreads();
    for (int idx = t; idx < 4 * 2 * 32 * 16; idx += 256) {
        int n = idx >> 10, rem = idx & 1023;
        int kc = rem >> 9, r2 = rem & 511, lane = r2 >> 4, e = r2 & 15;
        int o = n * 16 + (lane & 15);
        int k = kc * 32 + ((lane >> 4) << 4) + e;
        Wb[W_A1_OFF + idx] = (_Float16)(m1_w[k * 64 + o] * sc[k] * dd[o]);
    }
    __syncthreads();

    // --- layer 2 : in=64, out=3 (padded to 16 output rows) ---
    if (t < 64) {
        float a = a2_b[t];
        for (int j = 0; j < SDIM; ++j) a += ssm[j] * a2_w[j * 64 + t];
        sc[t] = a;
    }
    __syncthreads();
    if (t < 3) {
        float q = 0.f;
        for (int k = 0; k < 64; ++k) { float w = m2_w[k * 3 + t] * sc[k]; q += w * w; }
        dd[t] = rsqrtf(q + 1e-8f);
    }
    __syncthreads();
    for (int idx = t; idx < 2 * 32 * 16; idx += 256) {
        int kc = idx >> 9, r2 = idx & 511, lane = r2 >> 4, e = r2 & 15;
        int o = lane & 15;
        int k = kc * 32 + ((lane >> 4) << 4) + e;
        float v = 0.f;
        if (o < 3) v = m2_w[k * 3 + o] * sc[k] * dd[o];
        Wb[W_A2_OFF + idx] = (_Float16)v;
    }
}

// ---------------------------------------------------------------------------
// Kernel 3: fused hash-grid lookup + WMMA MLP, computed transposed:
//   D[o, m] = Wt(A) x featT(B) + bias(C)
// Weights stay in LDS and are re-read each tile with ds_load_b128; an opaque
// zero offset (empty asm) stops LICM from hoisting them into 112 VGPRs.
// ---------------------------------------------------------------------------
__global__ __launch_bounds__(256) void ngp_main_kernel(
    const float* __restrict__ base_tables,   // [LVLS][TSIZE][2]
    const _Float16* __restrict__ Wg,         // [BATCH][W_TOTAL]
    const float* __restrict__ bias0,         // [64]
    const float* __restrict__ bias1,         // [64]
    const float* __restrict__ bias2,         // [3]
    float* __restrict__ out)                 // [BATCH][3][NPIX]
{
    __shared__ __align__(32) _Float16 lds_w[W_TOTAL];        // 14 KB weights
    __shared__ __align__(32) _Float16 lds_feat[8][16][FEAT]; // per-wave [m][k]
    __shared__ __align__(32) _Float16 lds_h[8][16][HID];     // per-wave [m][o]

    const int b    = blockIdx.y;
    const int tid  = threadIdx.x;
    const int wave = tid >> 5;
    const int lane = tid & 31;
    const int m    = lane & 15;
    const int hi   = lane >> 4;

    // ---- stage this sample's packed weights into LDS (once per block) ----
    {
        const uint4* src = (const uint4*)(Wg + b * W_TOTAL);
        uint4* dst = (uint4*)lds_w;
        for (int i = tid; i < W_TOTAL * 2 / 16; i += 256) dst[i] = src[i];
    }
    __syncthreads();

    // ---- bias into C-accumulator layout (row o = n*16 + hi*8 + r) ----
    v8f cb0[4], cb1[4], cb2;
#pragma unroll
    for (int n = 0; n < 4; ++n)
#pragma unroll
        for (int r = 0; r < 8; ++r) {
            cb0[n][r] = bias0[n * 16 + hi * 8 + r];
            cb1[n][r] = bias1[n * 16 + hi * 8 + r];
        }
#pragma unroll
    for (int r = 0; r < 8; ++r) {
        int o = hi * 8 + r;
        cb2[r] = (o < 3) ? bias2[o] : 0.f;
    }

    // ---- per-lane level set (lev = 2i + hi): hoist resolution math ----
    float rm1[8];
#pragma unroll
    for (int i = 0; i < 8; ++i) {
        const int lev = 2 * i + hi;
        rm1[i] = floorf(16.f * exp2f((float)lev * (4.0f / 15.0f))) - 1.f;
    }

    const int PIX_PER_BLOCK = NPIX / gridDim.x;          // 1024 for grid.x=64
    const int ITERS = PIX_PER_BLOCK / 128;               // 8 waves * 16 px

    // Opaque zero: always 0 at runtime, unknowable to the optimizer, so the
    // per-tile LDS weight loads below cannot be hoisted out of the loop.
    int zoff = 0;

    for (int it = 0; it < ITERS; ++it) {
        asm volatile("" : "+v"(zoff));                   // no code emitted
        const _Float16* wbase = lds_w + zoff;            // stays addrspace(3)

        const int tile_base = blockIdx.x * PIX_PER_BLOCK + it * 128 + wave * 16;
        const int pix = tile_base + m;
        const float cx = ((float)(pix & 255) + 0.5f) * (1.f / 256.f);
        const float cy = ((float)(pix >> 8) + 0.5f) * (1.f / 256.f);

        // ---- hash-grid gather: 8 levels per lane (pixel m, level 2i+hi) ----
#pragma unroll
        for (int i = 0; i < 8; ++i) {
            const int lev = 2 * i + hi;
            const float sx = cx * rm1[i], sy = cy * rm1[i];
            const float fx = floorf(sx), fy = floorf(sy);
            const float tx = sx - fx, ty = sy - fy;
            const unsigned x0 = (unsigned)fx, y0 = (unsigned)fy;
            float f0 = 0.f, f1 = 0.f;
#pragma unroll
            for (int c = 0; c < 4; ++c) {
                const unsigned xc = x0 + (c >> 1), yc = y0 + (c & 1);
                const float w = ((c >> 1) ? tx : 1.f - tx) * ((c & 1) ? ty : 1.f - ty);
                const unsigned idx = (xc ^ (yc * 2654435761u)) & (TSIZE - 1);
                const float2 tv = *((const float2*)base_tables + (lev << 14) + idx);
                f0 += w * tv.x; f1 += w * tv.y;
            }
            v2h p; p[0] = (_Float16)f0; p[1] = (_Float16)f1;
            *(v2h*)&lds_feat[wave][m][2 * lev] = p;      // ds_store_b32
        }
        asm volatile("s_wait_dscnt 0x0" ::: "memory");

        // ---- layer 0 B operand: featT, lane = pixel col m, k = hi*16+e ----
        v16h bf;
        {
            const v8h* p = (const v8h*)&lds_feat[wave][m][hi * 16];
            v8h lo = p[0], hv = p[1];
            bf = __builtin_shufflevector(lo, hv, 0,1,2,3,4,5,6,7,8,9,10,11,12,13,14,15);
        }

        v8f acc[4];
#pragma unroll
        for (int n = 0; n < 4; ++n) {
            const v8h* p = (const v8h*)(wbase + W_A0_OFF + (n * 32 + lane) * 16);
            v8h lo = p[0], hv = p[1];
            v16h aop = __builtin_shufflevector(lo, hv, 0,1,2,3,4,5,6,7,8,9,10,11,12,13,14,15);
            acc[n] = __builtin_amdgcn_wmma_f32_16x16x32_f16(
                false, aop, false, bf, (short)0, cb0[n], false, false);
        }

        // ---- activation, pack to LDS [m][o] (one b128 per tile) ----
#pragma unroll
        for (int n = 0; n < 4; ++n) {
            v8h hh;
#pragma unroll
            for (int r = 0; r < 8; ++r) hh[r] = (_Float16)lrelu(acc[n][r]);
            *(v8h*)&lds_h[wave][m][n * 16 + hi * 8] = hh;
        }
        asm volatile("s_wait_dscnt 0x0" ::: "memory");

        // ---- layer 1: K=64 in two chunks ----
        v16h bh[2];
#pragma unroll
        for (int kc = 0; kc < 2; ++kc) {
            const v8h* p = (const v8h*)&lds_h[wave][m][kc * 32 + hi * 16];
            v8h lo = p[0], hv = p[1];
            bh[kc] = __builtin_shufflevector(lo, hv, 0,1,2,3,4,5,6,7,8,9,10,11,12,13,14,15);
        }
        v8f acc1[4];
#pragma unroll
        for (int n = 0; n < 4; ++n) {
            acc1[n] = cb1[n];
#pragma unroll
            for (int kc = 0; kc < 2; ++kc) {
                const v8h* p = (const v8h*)(wbase + W_A1_OFF + ((n * 2 + kc) * 32 + lane) * 16);
                v8h lo = p[0], hv = p[1];
                v16h aop = __builtin_shufflevector(lo, hv, 0,1,2,3,4,5,6,7,8,9,10,11,12,13,14,15);
                acc1[n] = __builtin_amdgcn_wmma_f32_16x16x32_f16(
                    false, aop, false, bh[kc], (short)0, acc1[n], false, false);
            }
        }
#pragma unroll
        for (int n = 0; n < 4; ++n) {
            v8h hh;
#pragma unroll
            for (int r = 0; r < 8; ++r) hh[r] = (_Float16)lrelu(acc1[n][r]);
            *(v8h*)&lds_h[wave][m][n * 16 + hi * 8] = hh;
        }
        asm volatile("s_wait_dscnt 0x0" ::: "memory");

        // ---- layer 2: out=3 (padded tile), bias in C, no activation ----
        v8f acc2 = cb2;
#pragma unroll
        for (int kc = 0; kc < 2; ++kc) {
            const v8h* p = (const v8h*)&lds_h[wave][m][kc * 32 + hi * 16];
            v8h blo = p[0], bhv = p[1];
            v16h bop = __builtin_shufflevector(blo, bhv, 0,1,2,3,4,5,6,7,8,9,10,11,12,13,14,15);
            const v8h* wp = (const v8h*)(wbase + W_A2_OFF + (kc * 32 + lane) * 16);
            v8h lo = wp[0], hv = wp[1];
            v16h aop = __builtin_shufflevector(lo, hv, 0,1,2,3,4,5,6,7,8,9,10,11,12,13,14,15);
            acc2 = __builtin_amdgcn_wmma_f32_16x16x32_f16(
                false, aop, false, bop, (short)0, acc2, false, false);
        }

        if (hi == 0) {
#pragma unroll
            for (int r = 0; r < 3; ++r)
                out[(b * 3 + r) * NPIX + pix] = acc2[r];
        }
    }
}

// ---------------------------------------------------------------------------
extern "C" void kernel_launch(void* const* d_in, const int* in_sizes, int n_in,
                              void* d_out, int out_size, void* d_ws, size_t ws_size,
                              hipStream_t stream) {
    (void)in_sizes; (void)n_in; (void)out_size; (void)ws_size;
    const float* z    = (const float*)d_in[0];
    const float* w1   = (const float*)d_in[1];  const float* b1 = (const float*)d_in[2];
    const float* w2   = (const float*)d_in[3];  const float* b2 = (const float*)d_in[4];
    const float* w3   = (const float*)d_in[5];  const float* b3 = (const float*)d_in[6];
    const float* base_tables = (const float*)d_in[7];
    const float* gen_w = (const float*)d_in[8];
    const float* a0_w = (const float*)d_in[9];  const float* a0_b = (const float*)d_in[10];
    const float* m0_w = (const float*)d_in[11]; const float* m0_b = (const float*)d_in[12];
    const float* a1_w = (const float*)d_in[13]; const float* a1_b = (const float*)d_in[14];
    const float* m1_w = (const float*)d_in[15]; const float* m1_b = (const float*)d_in[16];
    const float* a2_w = (const float*)d_in[17]; const float* a2_b = (const float*)d_in[18];
    const float* m2_w = (const float*)d_in[19]; const float* m2_b = (const float*)d_in[20];

    char* ws = (char*)d_ws;
    float*    s_ws = (float*)(ws + 0);                 // 8 KB
    _Float16* Wpk  = (_Float16*)(ws + 8192);           // 112 KB packed weights

    style_kernel<<<dim3(BATCH), dim3(256), 0, stream>>>(z, w1, b1, w2, b2, w3, b3, s_ws);
    prep_kernel<<<dim3(BATCH), dim3(256), 0, stream>>>(
        s_ws, gen_w, a0_w, a0_b, m0_w, a1_w, a1_b, m1_w, a2_w, a2_b, m2_w, Wpk);
    ngp_main_kernel<<<dim3(64, BATCH), dim3(256), 0, stream>>>(
        base_tables, Wpk, m0_b, m1_b, m2_b, (float*)d_out);
}
